// DCGRUCell_57062935495047
// MI455X (gfx1250) — matrix operand order repaired and checked by reading, not compiled, not run on hardware
//
#include <hip/hip_runtime.h>
#include <math.h>

// ---- types ---------------------------------------------------------------
typedef __bf16 bf16;
typedef __bf16 bfx8  __attribute__((ext_vector_type(8)));
typedef __bf16 v16bf __attribute__((ext_vector_type(16)));
typedef float  v8f   __attribute__((ext_vector_type(8)));

// ---- problem constants ---------------------------------------------------
#define NN  2048        // nodes
#define BBATCH 32       // batch
#define UU  64          // units
#define CC  66          // D + U
#define BC  2112        // BBATCH * CC
#define HXS 131072      // NN * UU

// ---- optional CDNA5 async global->LDS path --------------------------------
#if defined(__HIP_DEVICE_COMPILE__) &&                                   \
    __has_builtin(__builtin_amdgcn_global_load_async_to_lds_b128) &&     \
    __has_builtin(__builtin_amdgcn_s_wait_asynccnt)
#define USE_ASYNC_LDS 1
#else
#define USE_ASYNC_LDS 0
#endif

#if USE_ASYNC_LDS
// Builtin prototype (from clang diagnostic):
//   void (v4i addrspace(1)* src, v4i addrspace(3)* dst, imm int off, imm int cpol)
typedef int intx4 __attribute__((__vector_size__(16)));
typedef __attribute__((address_space(1))) intx4 g_intx4;
typedef __attribute__((address_space(3))) intx4 l_intx4;
// generic->AS1: VA is identical; generic->AS3: low 32 bits are the LDS offset.
__device__ __forceinline__ void async_cp16(const void* g, void* l) {
  __builtin_amdgcn_global_load_async_to_lds_b128(
      (g_intx4*)(uintptr_t)g,
      (l_intx4*)(uintptr_t)(uint32_t)(uintptr_t)l, 0, 0);
}
#endif

// Assemble a v16bf WMMA fragment from two 16-byte LDS chunks.
__device__ __forceinline__ v16bf frag2(const bf16* plo, const bf16* phi) {
  bfx8 lo = *(const bfx8*)plo;
  bfx8 hi = *(const bfx8*)phi;
  return __builtin_shufflevector(lo, hi, 0,1,2,3,4,5,6,7,8,9,10,11,12,13,14,15);
}

// ---- pack / convert kernels ---------------------------------------------
__global__ __launch_bounds__(256)
void k_cvt_bf16(const float* __restrict__ s, bf16* __restrict__ d, int n) {
  int i = blockIdx.x * 256 + threadIdx.x;
  if (i < n) d[i] = (bf16)s[i];
}

// Builds x0 in both layouts:
//   row-major X0[n][b*66 + c]      (projection operand)
//   transposed X0T[(b*66+c)][n]    (diffusion B operand)
// and seeds the input columns of the stage-2 buffers.
__global__ __launch_bounds__(256)
void k_pack_x0(const float* __restrict__ inp, const float* __restrict__ hx,
               bf16* __restrict__ X0, bf16* __restrict__ X0T,
               bf16* __restrict__ X0p, bf16* __restrict__ X0pT) {
  int idx = blockIdx.x * 256 + threadIdx.x;   // 0..65535 -> (n, b)
  int n = idx >> 5, b = idx & 31;
  size_t dst = (size_t)n * BC + (size_t)b * CC;
  size_t tcol = (size_t)b * CC;               // column index base in XT
  bf16 i0 = (bf16)inp[(size_t)b * (NN * 2) + n * 2 + 0];
  bf16 i1 = (bf16)inp[(size_t)b * (NN * 2) + n * 2 + 1];
  X0[dst] = i0;  X0[dst + 1] = i1;
  X0p[dst] = i0; X0p[dst + 1] = i1;
  X0T[(tcol + 0) * NN + n] = i0;  X0T[(tcol + 1) * NN + n] = i1;
  X0pT[(tcol + 0) * NN + n] = i0; X0pT[(tcol + 1) * NN + n] = i1;
  #pragma unroll 8
  for (int j = 0; j < UU; j++) {
    bf16 h = (bf16)hx[(size_t)b * HXS + n * UU + j];
    X0[dst + 2 + j] = h;
    X0T[(tcol + 2 + j) * NN + n] = h;
  }
}

// Wt[m][o][c] = W[c*5 + m][o] for c < 66 else 0  (zero-pad K 66 -> 96).
__global__ __launch_bounds__(256)
void k_pack_w(const float* __restrict__ W, bf16* __restrict__ Wt, int outn) {
  int i = blockIdx.x * 256 + threadIdx.x;
  int total = 5 * outn * 96;
  if (i >= total) return;
  int c = i % 96;
  int o = (i / 96) % outn;
  int m = i / (96 * outn);
  float v = (c < CC) ? W[(size_t)(c * 5 + m) * outn + o] : 0.f;
  Wt[i] = (bf16)v;
}

// ---- diffusion GEMM -------------------------------------------------------
// Y = alpha * (S @ X) + beta * Z   (bf16 in/out, f32 accumulate)
// S   : [2048][2048] row-major bf16
// XT,ZT,YT : transposed feature blocks [2112 cols][2048 nodes] bf16
// Yrm : row-major [2048][2112] bf16 (projection view), always written
// Block: 128x64 output tile, BK=32, 256 threads = 8 waves,
// wave owns a 2x2 grid of 16x16 WMMA tiles. Double-buffered LDS.
__global__ __launch_bounds__(256)
void k_diffuse(const bf16* __restrict__ S, const bf16* __restrict__ XT,
               const bf16* ZT, bf16* YT, bf16* __restrict__ Yrm,
               float alpha, float beta) {
  constexpr int LDA = 56;   // 112B rows: 16B aligned, conflict-clean
  constexpr int LDX = 40;   // 80B rows : 16B aligned, conflict-clean
  __shared__ bf16 As[2][128 * LDA];
  __shared__ bf16 Xs[2][64 * LDX];

  const int tid  = threadIdx.x;
  const int lane = tid & 31;
  const int wave = tid >> 5;
  const int lm   = lane & 15;
  const int hi   = lane >> 4;
  const int rt0  = (wave & 3) * 2;    // row tiles rt0, rt0+1
  const int ct0  = (wave >> 2) * 2;   // col tiles ct0, ct0+1
  const int colBase = blockIdx.x * 64;
  const int rowBase = blockIdx.y * 128;

  // staging indices: A tile 128x32 (2x16B per thread), X tile 64x32 (1x16B)
  const int am = tid >> 2;            // 0..63  (second chunk: +64)
  const int ac = (tid & 3) * 8;       // 0,8,16,24
  const int xc = tid >> 2;            // 0..63 column
  const int xg = (tid & 3) * 8;       // k group

  v8f acc[2][2];
  #pragma unroll
  for (int i = 0; i < 2; i++)
    #pragma unroll
    for (int j = 0; j < 2; j++) acc[i][j] = (v8f){0,0,0,0,0,0,0,0};

  // ---- prologue: stage chunk 0 into buffer 0
  {
#if USE_ASYNC_LDS
    async_cp16(&S[(size_t)(rowBase + am) * NN + ac],        &As[0][am * LDA + ac]);
    async_cp16(&S[(size_t)(rowBase + am + 64) * NN + ac],   &As[0][(am + 64) * LDA + ac]);
    async_cp16(&XT[(size_t)(colBase + xc) * NN + xg],       &Xs[0][xc * LDX + xg]);
    __builtin_amdgcn_s_wait_asynccnt(0);
#else
    *(bfx8*)&As[0][am * LDA + ac] =
        *(const bfx8*)&S[(size_t)(rowBase + am) * NN + ac];
    *(bfx8*)&As[0][(am + 64) * LDA + ac] =
        *(const bfx8*)&S[(size_t)(rowBase + am + 64) * NN + ac];
    *(bfx8*)&Xs[0][xc * LDX + xg] =
        *(const bfx8*)&XT[(size_t)(colBase + xc) * NN + xg];
#endif
    __syncthreads();
  }

  for (int kci = 0; kci < NN / 32; kci++) {
    const int cur = kci & 1;
    const int nxt = cur ^ 1;
    const bool more = (kci < NN / 32 - 1);
#if !USE_ASYNC_LDS
    bfx8 ra0{}, ra1{}, rx{};
#endif
    if (more) {
      const int kc = (kci + 1) * 32;
#if USE_ASYNC_LDS
      async_cp16(&S[(size_t)(rowBase + am) * NN + kc + ac],      &As[nxt][am * LDA + ac]);
      async_cp16(&S[(size_t)(rowBase + am + 64) * NN + kc + ac], &As[nxt][(am + 64) * LDA + ac]);
      async_cp16(&XT[(size_t)(colBase + xc) * NN + kc + xg],     &Xs[nxt][xc * LDX + xg]);
#else
      ra0 = *(const bfx8*)&S[(size_t)(rowBase + am) * NN + kc + ac];
      ra1 = *(const bfx8*)&S[(size_t)(rowBase + am + 64) * NN + kc + ac];
      rx  = *(const bfx8*)&XT[(size_t)(colBase + xc) * NN + kc + xg];
#endif
    }

    // ---- compute from buffer `cur`
    const bf16* Ab = As[cur];
    const bf16* Bb = Xs[cur];
    const bf16* ap0 = &Ab[(rt0 * 16 + lm) * LDA + 8 * hi];
    const bf16* ap1 = &Ab[((rt0 + 1) * 16 + lm) * LDA + 8 * hi];
    v16bf a0 = frag2(ap0, ap0 + 16);
    v16bf a1 = frag2(ap1, ap1 + 16);
    const bf16* bp0 = &Bb[(ct0 * 16 + lm) * LDX + 8 * hi];
    const bf16* bp1 = &Bb[((ct0 + 1) * 16 + lm) * LDX + 8 * hi];
    v16bf b0 = frag2(bp0, bp0 + 16);
    v16bf b1 = frag2(bp1, bp1 + 16);
    acc[0][0] = __builtin_amdgcn_wmma_f32_16x16x32_bf16(false, a0, false, b0,
                                                        (short)0, acc[0][0], false, false);
    acc[0][1] = __builtin_amdgcn_wmma_f32_16x16x32_bf16(false, a0, false, b1,
                                                        (short)0, acc[0][1], false, false);
    acc[1][0] = __builtin_amdgcn_wmma_f32_16x16x32_bf16(false, a1, false, b0,
                                                        (short)0, acc[1][0], false, false);
    acc[1][1] = __builtin_amdgcn_wmma_f32_16x16x32_bf16(false, a1, false, b1,
                                                        (short)0, acc[1][1], false, false);

    if (more) {
#if USE_ASYNC_LDS
      __builtin_amdgcn_s_wait_asynccnt(0);
#else
      *(bfx8*)&As[nxt][am * LDA + ac]        = ra0;
      *(bfx8*)&As[nxt][(am + 64) * LDA + ac] = ra1;
      *(bfx8*)&Xs[nxt][xc * LDX + xg]        = rx;
#endif
    }
    __syncthreads();
  }

  // ---- epilogue -----------------------------------------------------------
  const bool hasZ = (ZT != nullptr);
  const bool hasT = (YT != nullptr);
  #pragma unroll
  for (int i = 0; i < 2; i++) {
    const int row0 = rowBase + (rt0 + i) * 16 + hi * 8;   // 8 consecutive rows
    #pragma unroll
    for (int j = 0; j < 2; j++) {
      const int c = colBase + (ct0 + j) * 16 + lm;        // this lane's column
      v8f a = acc[i][j];
      bfx8 yv;
      if (hasZ) {
        bfx8 zv = *(const bfx8*)&ZT[(size_t)c * NN + row0];
        #pragma unroll
        for (int v = 0; v < 8; v++)
          yv[v] = (bf16)(alpha * a[v] + beta * (float)zv[v]);
      } else {
        #pragma unroll
        for (int v = 0; v < 8; v++) yv[v] = (bf16)(alpha * a[v]);
      }
      if (hasT) *(bfx8*)&YT[(size_t)c * NN + row0] = yv;  // transposed: 1x b128
      #pragma unroll
      for (int v = 0; v < 8; v++)                         // row-major view
        Yrm[(size_t)(row0 + v) * BC + c] = yv[v];
    }
  }
}

// ---- projection + activation ---------------------------------------------
// rows r = n*32 + b over the contiguous [65536, 66] row-major view of Xm.
// OUTN=128: sigmoid; o<64 -> r*hx into X0p (rm + transposed), o>=64 -> u_buf.
// OUTN=64 : tanh; fused new_state = u*hx + (1-u)*c into out.
template <int OUTN>
__global__ __launch_bounds__(256)
void k_proj(const bf16* __restrict__ X0, const bf16* __restrict__ X1,
            const bf16* __restrict__ X2, const bf16* __restrict__ X3,
            const bf16* __restrict__ X4,
            const bf16* __restrict__ Wt,   // [5][OUTN][96] bf16 pre-transposed
            const float* __restrict__ bias,
            const float* __restrict__ hx,
            float* u_buf, bf16* X0p, bf16* X0pT, float* out) {
  constexpr int NT  = OUTN / 32;
  constexpr int LDA = 40;
  constexpr int LDW = 104;
  __shared__ bf16 As[64 * LDA];
  __shared__ bf16 Ws[OUTN * LDW];

  const int tid  = threadIdx.x;
  const int lane = tid & 31;
  const int wave = tid >> 5;
  const int tm   = wave & 3;
  const int half = wave >> 2;
  const int lm   = lane & 15;
  const int hi   = lane >> 4;
  const int rowBase = blockIdx.x * 64;

  v8f acc[NT];
  #pragma unroll
  for (int i = 0; i < NT; i++) acc[i] = (v8f){0,0,0,0,0,0,0,0};

  const int ar = tid >> 2;
  const int ac = (tid & 3) * 8;
  const bf16* Xm[5] = {X0, X1, X2, X3, X4};

  #pragma unroll
  for (int m = 0; m < 5; m++) {
    const bf16* Xsrc = Xm[m];
    for (int cid = tid; cid < OUTN * 12; cid += 256) {
      int o = cid / 12, g = (cid % 12) * 8;
      *(bfx8*)&Ws[o * LDW + g] =
          *(const bfx8*)&Wt[((size_t)m * OUTN + o) * 96 + g];
    }
    __syncthreads();
    #pragma unroll
    for (int kc = 0; kc < 3; kc++) {
      bfx8 av;
      #pragma unroll
      for (int i = 0; i < 8; i++) {
        int c = kc * 32 + ac + i;
        av[i] = (c < CC) ? Xsrc[(size_t)(rowBase + ar) * CC + c] : (bf16)0.f;
      }
      *(bfx8*)&As[ar * LDA + ac] = av;
      __syncthreads();
      const bf16* ap = &As[(tm * 16 + lm) * LDA + 8 * hi];
      v16bf a = frag2(ap, ap + 16);
      #pragma unroll
      for (int i = 0; i < NT; i++) {
        int tn = half * NT + i;
        const bf16* bp = &Ws[(tn * 16 + lm) * LDW + kc * 32 + 8 * hi];
        v16bf b = frag2(bp, bp + 16);
        acc[i] = __builtin_amdgcn_wmma_f32_16x16x32_bf16(false, a, false, b,
                                                         (short)0, acc[i],
                                                         false, false);
      }
      __syncthreads();
    }
  }

  #pragma unroll
  for (int i = 0; i < NT; i++) {
    int o = (half * NT + i) * 16 + lm;
    float bv = bias[o];
    #pragma unroll
    for (int v = 0; v < 8; v++) {
      int r = rowBase + tm * 16 + hi * 8 + v;
      int n = r >> 5, b = r & 31;
      float val = acc[i][v] + bv;
      if (OUTN == 128) {
        float s = 1.f / (1.f + __expf(-val));               // sigmoid
        if (o < UU) {                                       // r-gate -> r*hx
          bf16 rh = (bf16)(s * hx[(size_t)b * HXS + n * UU + o]);
          X0p[(size_t)n * BC + (size_t)b * CC + 2 + o] = rh;
          X0pT[((size_t)b * CC + 2 + o) * NN + n] = rh;
        } else {                                            // u-gate
          u_buf[(size_t)r * UU + (o - UU)] = s;
        }
      } else {
        float cv = tanhf(val);
        float u  = u_buf[(size_t)r * UU + o];
        float h  = hx[(size_t)b * HXS + n * UU + o];
        out[(size_t)b * HXS + n * UU + o] = u * h + (1.f - u) * cv;
      }
    }
  }
}

// ---- launcher -------------------------------------------------------------
extern "C" void kernel_launch(void* const* d_in, const int* in_sizes, int n_in,
                              void* d_out, int out_size, void* d_ws, size_t ws_size,
                              hipStream_t stream) {
  (void)in_sizes; (void)n_in; (void)out_size; (void)ws_size;
  const float* inputs = (const float*)d_in[0];
  const float* hx     = (const float*)d_in[1];
  const float* s0     = (const float*)d_in[2];
  const float* s1     = (const float*)d_in[3];
  const float* W_ru   = (const float*)d_in[4];
  const float* b_ru   = (const float*)d_in[5];
  const float* W_c    = (const float*)d_in[6];
  const float* b_c    = (const float*)d_in[7];
  float* out = (float*)d_out;

  char* ws = (char*)d_ws;
  size_t off = 0;
  auto take = [&](size_t bytes) {
    size_t r = off;
    off += (bytes + 255) & ~(size_t)255;
    return r;
  };
  const size_t SZ_S = (size_t)NN * NN * sizeof(bf16);
  const size_t SZ_X = (size_t)NN * BC * sizeof(bf16);
  bf16* S0b  = (bf16*)(ws + take(SZ_S));
  bf16* S1b  = (bf16*)(ws + take(SZ_S));
  bf16* X0   = (bf16*)(ws + take(SZ_X));   // row-major views (projection)
  bf16* XA   = (bf16*)(ws + take(SZ_X));
  bf16* XB   = (bf16*)(ws + take(SZ_X));
  bf16* XC   = (bf16*)(ws + take(SZ_X));
  bf16* XD   = (bf16*)(ws + take(SZ_X));
  bf16* X0p  = (bf16*)(ws + take(SZ_X));
  bf16* X0T  = (bf16*)(ws + take(SZ_X));   // transposed views (diffusion)
  bf16* XAT  = (bf16*)(ws + take(SZ_X));
  bf16* XCT  = (bf16*)(ws + take(SZ_X));
  bf16* X0pT = (bf16*)(ws + take(SZ_X));
  bf16* Wrut = (bf16*)(ws + take((size_t)5 * 128 * 96 * sizeof(bf16)));
  bf16* Wct  = (bf16*)(ws + take((size_t)5 * 64  * 96 * sizeof(bf16)));
  float* ubuf = (float*)(ws + take((size_t)NN * BBATCH * UU * sizeof(float)));

  // 1) converts / packing
  int nS = NN * NN;
  k_cvt_bf16<<<(nS + 255) / 256, 256, 0, stream>>>(s0, S0b, nS);
  k_cvt_bf16<<<(nS + 255) / 256, 256, 0, stream>>>(s1, S1b, nS);
  k_pack_x0<<<(NN * BBATCH) / 256, 256, 0, stream>>>(inputs, hx, X0, X0T, X0p, X0pT);
  k_pack_w<<<(5 * 128 * 96 + 255) / 256, 256, 0, stream>>>(W_ru, Wrut, 128);
  k_pack_w<<<(5 * 64 * 96 + 255) / 256, 256, 0, stream>>>(W_c, Wct, 64);

  dim3 gd(BC / 64, NN / 128);   // (33, 16)

  // 2) gconv_ru diffusion
  k_diffuse<<<gd, 256, 0, stream>>>(S0b, X0T, nullptr, XAT, XA, 1.f, 0.f);
  k_diffuse<<<gd, 256, 0, stream>>>(S0b, XAT, X0T, nullptr, XB, 2.f, -1.f);
  k_diffuse<<<gd, 256, 0, stream>>>(S1b, X0T, nullptr, XCT, XC, 1.f, 0.f);
  k_diffuse<<<gd, 256, 0, stream>>>(S1b, XCT, X0T, nullptr, XD, 2.f, -1.f);

  // 3) ru projection: sigmoid, r*hx -> x0' (both layouts), u -> u_buf
  k_proj<128><<<(NN * BBATCH) / 64, 256, 0, stream>>>(
      X0, XA, XB, XC, XD, Wrut, b_ru, hx, ubuf, X0p, X0pT, nullptr);

  // 4) gconv_c diffusion on x0'
  k_diffuse<<<gd, 256, 0, stream>>>(S0b, X0pT, nullptr, XAT, XA, 1.f, 0.f);
  k_diffuse<<<gd, 256, 0, stream>>>(S0b, XAT, X0pT, nullptr, XB, 2.f, -1.f);
  k_diffuse<<<gd, 256, 0, stream>>>(S1b, X0pT, nullptr, XCT, XC, 1.f, 0.f);
  k_diffuse<<<gd, 256, 0, stream>>>(S1b, XCT, X0pT, nullptr, XD, 2.f, -1.f);

  // 5) c projection: tanh + fused new_state into d_out
  k_proj<64><<<(NN * BBATCH) / 64, 256, 0, stream>>>(
      X0p, XA, XB, XC, XD, Wct, b_c, hx, ubuf, nullptr, nullptr, out);
}
